// GCNLSTMTransformer_72490458022115
// MI455X (gfx1250) — compile-verified, matrix operand order checked by reading, not run on hardware
//
#include <hip/hip_runtime.h>
#include <hip/hip_bf16.h>

// ---------------- model constants ----------------
#define NB     32          // batch
#define TT     417         // timesteps
#define VV     25          // joints
#define DD     2           // coords
#define VD     50          // V*D == LH
#define NODES  10425       // T*V
#define EE     80000       // interframe edges
#define ETOT   (EE + NODES)
#define HID    20
#define LH     50
#define NHD    5
#define HDIM   10          // LH/NHD
#define FF     2048
#define NL     3
#define CO     85
#define NC     6
#define NSTG   2
#define SZ     83          // int(T*0.2)
#define EPSF   1e-5f
#define ROWS   (NB * CO)   // 2720 transformer rows

typedef __attribute__((ext_vector_type(16))) _Float16 v16h;
typedef __attribute__((ext_vector_type(8)))  float    v8f;

// ---------------- utility kernels ----------------
__global__ void k_zero(float* p, long n) {
    long i = (long)blockIdx.x * blockDim.x + threadIdx.x;
    if (i < n) p[i] = 0.f;
}

// degree over dst (E edges + NODES self loops)
__global__ void k_deg(const int* ei, float* deg) {
    int e = blockIdx.x * blockDim.x + threadIdx.x;
    if (e < EE)        atomicAdd(&deg[ei[EE + e]], 1.0f);
    else if (e < ETOT) atomicAdd(&deg[e - EE], 1.0f);
}

__global__ void k_dinv(const float* deg, float* dinv) {
    int i = blockIdx.x * blockDim.x + threadIdx.x;
    if (i < NODES) dinv[i] = (deg[i] > 0.f) ? rsqrtf(deg[i]) : 0.f;
}

__global__ void k_norm(const int* ei, const float* dinv, float* norm) {
    int e = blockIdx.x * blockDim.x + threadIdx.x;
    if (e < EE)        norm[e] = dinv[ei[e]] * dinv[ei[EE + e]];
    else if (e < ETOT) { float d = dinv[e - EE]; norm[e] = d * d; }
}

// ---------------- LSTM: one block per batch sample, sequential over T ----------------
__global__ void k_lstm(const float* xin, const float* Wih, const float* Whh,
                       const float* bih, const float* bhh,
                       const float* h0, const float* c0, float* xout) {
    __shared__ float sh[LH], sc[LH], sx[LH], sg[4 * LH];
    int n = blockIdx.x, tid = threadIdx.x;
    if (tid < LH) { sh[tid] = h0[n * LH + tid]; sc[tid] = c0[n * LH + tid]; }
    __syncthreads();
    for (int t = 0; t < TT; ++t) {
        if (tid < LH) sx[tid] = xin[((long)n * TT + t) * VD + tid];
        __syncthreads();
        if (tid < 4 * LH) {
            float acc = bih[tid] + bhh[tid];
            for (int k = 0; k < LH; ++k)
                acc += sx[k] * Wih[k * 4 * LH + tid] + sh[k] * Whh[k * 4 * LH + tid];
            sg[tid] = acc;
        }
        __syncthreads();
        if (tid < LH) {
            float gi = sg[tid], gf = sg[LH + tid], gg = sg[2 * LH + tid], go = sg[3 * LH + tid];
            float si = 1.f / (1.f + expf(-gi));
            float sf = 1.f / (1.f + expf(-gf));
            float so = 1.f / (1.f + expf(-go));
            float c  = sf * sc[tid] + si * tanhf(gg);
            float h  = so * tanhf(c);
            sc[tid] = c; sh[tid] = h;
            xout[((long)n * TT + t) * VD + tid] = h;
        }
        __syncthreads();
    }
}

// ---------------- GCN layer 1 scatter: msg = norm * (X[src] @ W1), atomic into AGG ----------------
__global__ void k_gcn1_scatter(const float* X, const float* W1, const int* ei,
                               const float* norm, float* agg) {
    long tid = (long)blockIdx.x * blockDim.x + threadIdx.x;
    if (tid >= (long)ETOT * NB * HID) return;
    int c = (int)(tid % HID);
    long r = tid / HID;
    int n = (int)(r % NB);
    int e = (int)(r / NB);
    int s = (e < EE) ? ei[e]      : (e - EE);
    int d = (e < EE) ? ei[EE + e] : (e - EE);
    float w = norm[e];
    const float* xp = X + ((long)n * NODES + s) * DD;
    float msg = w * (xp[0] * W1[c] + xp[1] * W1[HID + c]);
    atomicAdd(&agg[((long)n * NODES + d) * HID + c], msg);
}

// add gcn1 bias in place, accumulate per-channel sum / sumsq (block partials)
__global__ void k_bias_bnstats(float* h, const float* b1, float* stats, long total) {
    __shared__ float ps[HID], pq[HID];
    int tid = threadIdx.x;
    if (tid < HID) { ps[tid] = 0.f; pq[tid] = 0.f; }
    __syncthreads();
    long i = (long)blockIdx.x * blockDim.x + tid;
    if (i < total) {
        int c = (int)(i % HID);
        float v = h[i] + b1[c];
        h[i] = v;
        atomicAdd(&ps[c], v);
        atomicAdd(&pq[c], v * v);
    }
    __syncthreads();
    if (tid < HID) { atomicAdd(&stats[tid], ps[tid]); atomicAdd(&stats[HID + tid], pq[tid]); }
}

__global__ void k_bn_relu(float* h, const float* stats, const float* g, const float* b, long total) {
    long i = (long)blockIdx.x * blockDim.x + threadIdx.x;
    if (i >= total) return;
    int c = (int)(i % HID);
    float cnt = (float)((long)NB * NODES);
    float m = stats[c] / cnt;
    float var = stats[HID + c] / cnt - m * m;
    float v = (h[i] - m) * rsqrtf(var + EPSF) * g[c] + b[c];
    h[i] = fmaxf(v, 0.f);
}

// ---------------- GCN layer 2 scatter: msg = norm * (H[src] @ W2) ----------------
__global__ void k_gcn2_scatter(const float* H, const float* W2, const int* ei,
                               const float* norm, float* agg2) {
    long tid = (long)blockIdx.x * blockDim.x + threadIdx.x;
    if (tid >= (long)ETOT * NB) return;
    int n = (int)(tid % NB);
    int e = (int)(tid / NB);
    int s = (e < EE) ? ei[e]      : (e - EE);
    int d = (e < EE) ? ei[EE + e] : (e - EE);
    float w = norm[e];
    const float* hp = H + ((long)n * NODES + s) * HID;
    float a0 = 0.f, a1 = 0.f;
    for (int c = 0; c < HID; ++c) { a0 += hp[c] * W2[c * DD]; a1 += hp[c] * W2[c * DD + 1]; }
    float* op = agg2 + ((long)n * NODES + d) * DD;
    atomicAdd(&op[0], w * a0);
    atomicAdd(&op[1], w * a1);
}

__global__ void k_gcn2_bias(const float* agg2, const float* b2, float* xout, long total) {
    long i = (long)blockIdx.x * blockDim.x + threadIdx.x;
    if (i < total) xout[i] = agg2[i] + b2[i % DD];
}

// ---------------- WMMA GEMM: C = A(M,K) @ B(K,N) [+bias][+relu] ----------------
// Block = 4 waves, 32x64 C macro-tile. Tiles are staged into LDS *pre-swizzled into
// WMMA fragment order*, so each wave's fragment read is one 32-byte contiguous
// v16h load (2x ds_load_b128) per operand. Staging is branch-free everywhere:
// interior tiles load unguarded; edge tiles gather all clamped loads into registers
// first (one batched wait), then select/convert/store. Each wave holds one A
// fragment and two B fragments -> 2 WMMAs per K-step.
// biasMode: 0 none, 1 per-column (N), 2 per-row (M)
#define BM 32
#define BN 64
#define BK 32

__global__ __launch_bounds__(128)
void k_wmma_gemm(const float* A, const float* B, const float* bias, float* C,
                 int M, int N, int K, long strideA, long strideB, long strideC,
                 int biasMode, int relu) {
    // fragment-order LDS: [subtile][lane][slot], one lane's 16 halves contiguous (32B)
    __shared__ alignas(32) _Float16 aF[2][32][16];
    __shared__ alignas(32) _Float16 bF[4][32][16];
    const float* Ab = A + (long)blockIdx.z * strideA;
    const float* Bb = B + (long)blockIdx.z * strideB;
    float*       Cb = C + (long)blockIdx.z * strideC;
    int tid  = threadIdx.x;
    int wave = tid >> 5;
    int lane = tid & 31;
    int half = lane >> 4, lm = lane & 15;
    int wMs = wave >> 1;                 // wave's M subtile (0/1)
    int wNs = wave & 1;                  // wave's first N subtile (0/1; second is +2)
    int m0 = blockIdx.x * BM;
    int n0 = blockIdx.y * BN;
    bool fullA = (m0 + BM <= M);
    bool fullB = (n0 + BN <= N);

    v8f acc0 = {}, acc1 = {};
    for (int k0 = 0; k0 < K; k0 += BK) {
        bool fullK = (k0 + BK <= K);
        // ---- stage A tile (BM x BK) into fragment order ----
        // element (r,c): lane = ((c>>3)&1)*16 + (r&15), slot = (c&7) + 8*(c>>4)
        if (fullK && fullA) {
#pragma unroll
            for (int i = 0; i < 8; ++i) {
                int linear = i * 128 + tid;          // 0..1023
                int r = linear >> 5, c = linear & 31;
                float v = Ab[(long)(m0 + r) * K + (k0 + c)];
                aF[r >> 4][((c >> 3) & 1) * 16 + (r & 15)][(c & 7) + 8 * (c >> 4)] = (_Float16)v;
            }
        } else {
            float va[8];
#pragma unroll
            for (int i = 0; i < 8; ++i) {            // gather phase: batched loads
                int linear = i * 128 + tid;
                int r = linear >> 5, c = linear & 31;
                int mm = m0 + r, kk = k0 + c;
                int mc = (mm < M) ? mm : (M - 1);
                int kc = (kk < K) ? kk : (K - 1);
                va[i] = Ab[(long)mc * K + kc];       // clamped: always in-bounds
            }
#pragma unroll
            for (int i = 0; i < 8; ++i) {            // select/convert/store phase
                int linear = i * 128 + tid;
                int r = linear >> 5, c = linear & 31;
                bool ok = (m0 + r < M) && (k0 + c < K);
                float v = ok ? va[i] : 0.f;
                aF[r >> 4][((c >> 3) & 1) * 16 + (r & 15)][(c & 7) + 8 * (c >> 4)] = (_Float16)v;
            }
        }
        // ---- stage B tile (BK x BN) into fragment order ----
        // element (r=k,c=n): sub = c>>4, lane = ((r>>3)&1)*16 + (c&15), slot = (r&7) + 8*(r>>4)
        if (fullK && fullB) {
#pragma unroll
            for (int i = 0; i < 16; ++i) {
                int linear = i * 128 + tid;          // 0..2047
                int r = linear >> 6, c = linear & 63;
                float v = Bb[(long)(k0 + r) * N + (n0 + c)];
                bF[c >> 4][((r >> 3) & 1) * 16 + (c & 15)][(r & 7) + 8 * (r >> 4)] = (_Float16)v;
            }
        } else {
            float vb[16];
#pragma unroll
            for (int i = 0; i < 16; ++i) {           // gather phase: batched loads
                int linear = i * 128 + tid;
                int r = linear >> 6, c = linear & 63;
                int kk = k0 + r, nn = n0 + c;
                int kc = (kk < K) ? kk : (K - 1);
                int nc = (nn < N) ? nn : (N - 1);
                vb[i] = Bb[(long)kc * N + nc];       // clamped: always in-bounds
            }
#pragma unroll
            for (int i = 0; i < 16; ++i) {           // select/convert/store phase
                int linear = i * 128 + tid;
                int r = linear >> 6, c = linear & 63;
                bool ok = (k0 + r < K) && (n0 + c < N);
                float v = ok ? vb[i] : 0.f;
                bF[c >> 4][((r >> 3) & 1) * 16 + (c & 15)][(r & 7) + 8 * (r >> 4)] = (_Float16)v;
            }
        }
        __syncthreads();
        // ---- fragment loads: contiguous 32B per lane -> wide ds loads ----
        v16h af  = *reinterpret_cast<const v16h*>(&aF[wMs][lane][0]);
        v16h bf0 = *reinterpret_cast<const v16h*>(&bF[wNs][lane][0]);
        v16h bf1 = *reinterpret_cast<const v16h*>(&bF[wNs + 2][lane][0]);
        acc0 = __builtin_amdgcn_wmma_f32_16x16x32_f16(false, af, false, bf0,
                                                      (short)0, acc0, false, false);
        acc1 = __builtin_amdgcn_wmma_f32_16x16x32_f16(false, af, false, bf1,
                                                      (short)0, acc1, false, false);
        __syncthreads();
    }
    // C/D layout: VGPR i, lanes0-15 -> M=i, lanes16-31 -> M=i+8; N = lane%16
    int cn0 = n0 + wNs * 16 + lm;
    int cn1 = cn0 + 32;
#pragma unroll
    for (int i = 0; i < 8; ++i) {
        int cm = m0 + wMs * 16 + i + 8 * half;
        if (cm < M) {
            if (cn0 < N) {
                float v = acc0[i];
                if (biasMode == 1) v += bias[cn0];
                else if (biasMode == 2) v += bias[cm];
                if (relu) v = fmaxf(v, 0.f);
                Cb[(long)cm * N + cn0] = v;
            }
            if (cn1 < N) {
                float v = acc1[i];
                if (biasMode == 1) v += bias[cn1];
                else if (biasMode == 2) v += bias[cm];
                if (relu) v = fmaxf(v, 0.f);
                Cb[(long)cm * N + cn1] = v;
            }
        }
    }
}

// ---------------- attention core: one block per (b, head) ----------------
__global__ void k_attn(const float* qkv, float* obuf) {
    __shared__ float sq[32][HDIM], sk[32][HDIM], sv[32][HDIM], satt[32][32];
    int b = blockIdx.x / NHD, h = blockIdx.x % NHD;
    int tid = threadIdx.x;
    if (tid < 32 * HDIM) {
        int s = tid / HDIM, d = tid % HDIM;
        long r = (long)(s * CO + b) * (3 * LH);
        sq[s][d] = qkv[r + h * HDIM + d];
        sk[s][d] = qkv[r + LH + h * HDIM + d];
        sv[s][d] = qkv[r + 2 * LH + h * HDIM + d];
    }
    __syncthreads();
    {   // scores: 32x32, 1024 threads
        int s = tid / 32, t = tid % 32;
        float acc = 0.f;
        for (int d = 0; d < HDIM; ++d) acc += sq[s][d] * sk[t][d];
        satt[s][t] = acc * rsqrtf((float)HDIM);
    }
    __syncthreads();
    if (tid < 32) {   // softmax over t
        float mx = -1e30f;
        for (int t = 0; t < 32; ++t) mx = fmaxf(mx, satt[tid][t]);
        float se = 0.f;
        for (int t = 0; t < 32; ++t) { float e = expf(satt[tid][t] - mx); satt[tid][t] = e; se += e; }
        float inv = 1.f / se;
        for (int t = 0; t < 32; ++t) satt[tid][t] *= inv;
    }
    __syncthreads();
    if (tid < 32 * HDIM) {
        int s = tid / HDIM, d = tid % HDIM;
        float acc = 0.f;
        for (int t = 0; t < 32; ++t) acc += satt[s][t] * sv[t][d];
        obuf[(long)(s * CO + b) * LH + h * HDIM + d] = acc;
    }
}

// ---------------- residual + layernorm, one thread per row of 50 ----------------
__global__ void k_add_ln(float* t, const float* a, const float* g, const float* b, int rows) {
    int r = blockIdx.x * blockDim.x + threadIdx.x;
    if (r >= rows) return;
    float* tr = t + (long)r * LH;
    const float* ar = a + (long)r * LH;
    float buf[LH];
    float m = 0.f;
    for (int i = 0; i < LH; ++i) { buf[i] = tr[i] + ar[i]; m += buf[i]; }
    m *= (1.f / LH);
    float v = 0.f;
    for (int i = 0; i < LH; ++i) { float d = buf[i] - m; v += d * d; }
    v *= (1.f / LH);
    float inv = rsqrtf(v + EPSF);
    for (int i = 0; i < LH; ++i) tr[i] = (buf[i] - m) * inv * g[i] + b[i];
}

// ---------------- loss1: MSE over sampled next-frame predictions ----------------
__global__ void k_loss1(const float* x1, const float* x, const int* idx2,
                        const float* fcW, const float* fcb, float* acc) {
    long tid = (long)blockIdx.x * blockDim.x + threadIdx.x;
    float val = 0.f;
    if (tid < (long)NB * SZ * LH) {
        int l = (int)(tid % LH);
        long r = tid / LH;
        int j = (int)(r % SZ);
        int n = (int)(r / SZ);
        int ti = idx2[n * SZ + j];
        const float* xr = x1 + ((long)n * TT + ti) * VD;
        float p = fcb[l];
        for (int k = 0; k < LH; ++k) p += xr[k] * fcW[k * LH + l];
        float tg = x[((long)n * TT + ti + 1) * VD + l];
        float d = p - tg;
        val = d * d;
    }
    __shared__ float sb[256];
    sb[threadIdx.x] = val;
    __syncthreads();
    for (int st = 128; st > 0; st >>= 1) {
        if (threadIdx.x < st) sb[threadIdx.x] += sb[threadIdx.x + st];
        __syncthreads();
    }
    if (threadIdx.x == 0) atomicAdd(acc, sb[0]);
}

// ---------------- final: softmax probs, loss2, finalize loss1 ----------------
__global__ void k_final(const float* logits, const int* y, const float* loss1acc, float* out) {
    __shared__ float terms[NB];
    int n = threadIdx.x;
    if (n < NB) {
        float l[NC];
        float mx = -1e30f;
        for (int c = 0; c < NC; ++c) { l[c] = logits[n * NC + c]; mx = fmaxf(mx, l[c]); }
        float se = 0.f;
        for (int c = 0; c < NC; ++c) se += expf(l[c] - mx);
        float logZ = mx + logf(se);
        for (int c = 0; c < NC; ++c) out[n * NC + c] = expf(l[c] - logZ);
        terms[n] = -(l[y[n]] - logZ);
    }
    __syncthreads();
    if (n == 0) {
        float s = 0.f;
        for (int i = 0; i < NB; ++i) s += terms[i];
        out[NB * NC]     = loss1acc[0] / (float)((long)NB * SZ * LH);  // loss1
        out[NB * NC + 1] = s / (float)NB;                              // loss2
    }
}

// ---------------- host driver ----------------
static inline unsigned nblk(long n, int b) { return (unsigned)((n + b - 1) / b); }

extern "C" void kernel_launch(void* const* d_in, const int* in_sizes, int n_in,
                              void* d_out, int out_size, void* d_ws, size_t ws_size,
                              hipStream_t stream) {
    // inputs in setup_inputs() order
    const float* x        = (const float*)d_in[0];
    const int*   y        = (const int*)  d_in[1];
    const int*   ei       = (const int*)  d_in[2];   // (2,E) flat
    const int*   idx2     = (const int*)  d_in[3];
    const float* h0s      = (const float*)d_in[4];
    const float* c0s      = (const float*)d_in[5];
    const float* lstm_Wih = (const float*)d_in[6];
    const float* lstm_Whh = (const float*)d_in[7];
    const float* lstm_bih = (const float*)d_in[8];
    const float* lstm_bhh = (const float*)d_in[9];
    const float* gcn1_W   = (const float*)d_in[10];
    const float* gcn1_b   = (const float*)d_in[11];
    const float* bn_g     = (const float*)d_in[12];
    const float* bn_b     = (const float*)d_in[13];
    const float* gcn2_W   = (const float*)d_in[14];
    const float* gcn2_b   = (const float*)d_in[15];
    const float* fc_W     = (const float*)d_in[16];
    const float* fc_b     = (const float*)d_in[17];
    const float* conv_W   = (const float*)d_in[18];
    const float* conv_b   = (const float*)d_in[19];
    const float* attn_inW = (const float*)d_in[20];
    const float* attn_inb = (const float*)d_in[21];
    const float* attn_outW= (const float*)d_in[22];
    const float* attn_outb= (const float*)d_in[23];
    const float* ff1W     = (const float*)d_in[24];
    const float* ff1b     = (const float*)d_in[25];
    const float* ff2W     = (const float*)d_in[26];
    const float* ff2b     = (const float*)d_in[27];
    const float* ln1g     = (const float*)d_in[28];
    const float* ln1b     = (const float*)d_in[29];
    const float* ln2g     = (const float*)d_in[30];
    const float* ln2b     = (const float*)d_in[31];
    const float* Wc1      = (const float*)d_in[32];
    const float* bc1      = (const float*)d_in[33];
    const float* Wc2      = (const float*)d_in[34];
    const float* bc2      = (const float*)d_in[35];
    float* out = (float*)d_out;

    // bump allocator over workspace
    char* base = (char*)d_ws;
    size_t off = 0;
    auto alloc = [&](long nfloats) -> float* {
        off = (off + 255) & ~(size_t)255;
        float* p = (float*)(base + off);
        off += (size_t)nfloats * sizeof(float);
        return p;
    };
    float* deg     = alloc(NODES);
    float* dinv    = alloc(NODES);
    float* norm    = alloc(ETOT);
    float* x1a     = alloc((long)NB * TT * VD);       // lstm output / gcn input
    float* x1b     = alloc((long)NB * TT * VD);       // gcn output / stage result
    float* agg1    = alloc((long)NB * NODES * HID);   // 26.7 MB
    float* agg2    = alloc((long)NB * NODES * DD);
    float* bnstats = alloc(2 * HID);
    float* l1acc   = alloc(1);
    float* tbuf    = alloc((long)ROWS * LH);          // transformer state
    float* qkv     = alloc((long)ROWS * 3 * LH);
    float* obuf    = alloc((long)ROWS * LH);          // attn concat heads
    float* abuf    = alloc((long)ROWS * LH);          // projected / ff2 result
    float* ffbuf   = alloc((long)ROWS * FF);          // 22.3 MB
    float* h1buf   = alloc((long)NB * LH);
    float* logits  = alloc((long)NB * NC);

    auto gemm = [&](const float* A, const float* B, const float* bias, float* C,
                    int M, int N, int K, long sA, long sB, long sC,
                    int batch, int biasMode, int relu) {
        dim3 g(nblk(M, BM), nblk(N, BN), batch);
        k_wmma_gemm<<<g, 128, 0, stream>>>(A, B, bias, C, M, N, K, sA, sB, sC, biasMode, relu);
    };

    // ---- graph normalization ----
    k_zero<<<nblk(NODES, 256), 256, 0, stream>>>(deg, NODES);
    k_deg<<<nblk(ETOT, 256), 256, 0, stream>>>(ei, deg);
    k_dinv<<<nblk(NODES, 256), 256, 0, stream>>>(deg, dinv);
    k_norm<<<nblk(ETOT, 256), 256, 0, stream>>>(ei, dinv, norm);

    // ---- stacked (LSTM -> GCN) stages ----
    const float* xin = x;
    for (int s = 0; s < NSTG; ++s) {
        k_lstm<<<NB, 256, 0, stream>>>(xin,
            lstm_Wih + (long)s * LH * 4 * LH, lstm_Whh + (long)s * LH * 4 * LH,
            lstm_bih + (long)s * 4 * LH,      lstm_bhh + (long)s * 4 * LH,
            h0s + (long)s * NB * LH,          c0s + (long)s * NB * LH, x1a);

        long tot1 = (long)NB * NODES * HID;
        k_zero<<<nblk(tot1, 256), 256, 0, stream>>>(agg1, tot1);
        k_gcn1_scatter<<<nblk((long)ETOT * NB * HID, 256), 256, 0, stream>>>(
            x1a, gcn1_W + (long)s * DD * HID, ei, norm, agg1);
        k_zero<<<1, 64, 0, stream>>>(bnstats, 2 * HID);
        k_bias_bnstats<<<nblk(tot1, 256), 256, 0, stream>>>(agg1, gcn1_b + (long)s * HID, bnstats, tot1);
        k_bn_relu<<<nblk(tot1, 256), 256, 0, stream>>>(agg1, bnstats,
            bn_g + (long)s * HID, bn_b + (long)s * HID, tot1);

        long tot2 = (long)NB * NODES * DD;
        k_zero<<<nblk(tot2, 256), 256, 0, stream>>>(agg2, tot2);
        k_gcn2_scatter<<<nblk((long)ETOT * NB, 256), 256, 0, stream>>>(
            agg1, gcn2_W + (long)s * HID * DD, ei, norm, agg2);
        k_gcn2_bias<<<nblk(tot2, 256), 256, 0, stream>>>(agg2, gcn2_b + (long)s * DD, x1b, tot2);
        xin = x1b;
    }

    // ---- loss1 (sampled next-frame MSE through fc) ----
    k_zero<<<1, 32, 0, stream>>>(l1acc, 1);
    k_loss1<<<nblk((long)NB * SZ * LH, 256), 256, 0, stream>>>(x1b, x, idx2, fc_W, fc_b, l1acc);

    // ---- conv1d (k=1, time-as-channels): batched WMMA GEMM (85,417)@(417,50) ----
    gemm(conv_W, x1b, conv_b, tbuf, CO, LH, TT,
         0, (long)TT * VD, (long)CO * LH, NB, /*bias per-row*/2, 0);

    // ---- transformer encoder (post-norm) ----
    for (int l = 0; l < NL; ++l) {
        gemm(tbuf, attn_inW + (long)l * LH * 3 * LH, attn_inb + (long)l * 3 * LH, qkv,
             ROWS, 3 * LH, LH, 0, 0, 0, 1, 1, 0);
        k_attn<<<CO * NHD, 1024, 0, stream>>>(qkv, obuf);
        gemm(obuf, attn_outW + (long)l * LH * LH, attn_outb + (long)l * LH, abuf,
             ROWS, LH, LH, 0, 0, 0, 1, 1, 0);
        k_add_ln<<<nblk(ROWS, 128), 128, 0, stream>>>(tbuf, abuf,
             ln1g + (long)l * LH, ln1b + (long)l * LH, ROWS);
        gemm(tbuf, ff1W + (long)l * LH * FF, ff1b + (long)l * FF, ffbuf,
             ROWS, FF, LH, 0, 0, 0, 1, 1, /*relu*/1);
        gemm(ffbuf, ff2W + (long)l * FF * LH, ff2b + (long)l * LH, abuf,
             ROWS, LH, FF, 0, 0, 0, 1, 1, 0);
        k_add_ln<<<nblk(ROWS, 128), 128, 0, stream>>>(tbuf, abuf,
             ln2g + (long)l * LH, ln2b + (long)l * LH, ROWS);
    }

    // ---- classifier ----
    gemm(tbuf, Wc1, bc1, h1buf, NB, LH, CO * LH, 0, 0, 0, 1, 1, /*relu*/1);
    gemm(h1buf, Wc2, bc2, logits, NB, NC, LH, 0, 0, 0, 1, 1, 0);

    // ---- probs + losses ----
    k_final<<<1, NB, 0, stream>>>(logits, y, l1acc, out);

    (void)in_sizes; (void)n_in; (void)out_size; (void)ws_size;
}
